// SSCSNN_15831249453651
// MI455X (gfx1250) — matrix operand order repaired and checked by reading, not compiled, not run on hardware
//
#include <hip/hip_runtime.h>

// MI455X (gfx1250, wave32) persistent spiking-RNN forward.
// GEMMs via v_wmma_f32_16x16x32_bf16; spikes kept in LDS (bf16), membrane
// state in VGPRs (WMMA C-layout). One workgroup owns a 16-row batch tile for
// all 250 time steps (recurrence is batch-row independent -> no global sync).
// Weights (and, workspace permitting, x) are pre-converted to bf16 in d_ws.

typedef __bf16 v16bf __attribute__((ext_vector_type(16)));
typedef __bf16 v8bf  __attribute__((ext_vector_type(8)));
typedef float  v8f   __attribute__((ext_vector_type(8)));
typedef float  v4f   __attribute__((ext_vector_type(4)));

#define SPIKE_TH 1.0f
#define BB 256
#define TT 250
#define CC 700
#define CP 704   // C padded to multiple of 64 (pad cols of W1b / xb are zero)
#define H1 1024
#define H2 512
#define OO 35
#define OP 48    // O padded to 3 WMMA tiles (pad rows of W3b are zero)

#define NWAVES 16
#define NT1 4    // H1 tiles per wave: 1024 / (16 waves * 16 cols)
#define NT2 2    // H2 tiles per wave:  512 / (16 waves * 16 cols)

// element stride between consecutive N-tiles (16 weight rows); compile-time
// constants so tile addressing folds into the 24-bit instruction offset.
#define W1_TILE (16 * CP)   // 11264 elem = 22528 B
#define WR_TILE (16 * H1)   // 16384 elem = 32768 B
#define W2_TILE (16 * H1)

// workspace layout (bytes)
#define WS_W1   0u
#define WS_WREC 1441792u                       // + 1024*704*2
#define WS_W2   (WS_WREC + 2097152u)           // + 1024*1024*2
#define WS_W3   (WS_W2 + 1048576u)             // +  512*1024*2
#define WS_XB   (WS_W3 + 49152u)               // +   48*512*2
#define WS_NEED ((size_t)WS_XB + (size_t)BB * TT * CP * 2u)

__device__ __forceinline__ v8f wmma_bf16(v16bf a, v16bf b, v8f c) {
  return __builtin_amdgcn_wmma_f32_16x16x32_bf16(
      /*neg_a=*/false, a, /*neg_b=*/false, b,
      /*c_mod=*/(short)0, c, /*reuse_a=*/false, /*reuse_b=*/false);
}

// WGP-scope (scope 0) prefetch: pulls the line into all cache levels, stops
// on WGP$ hit (the __builtin_prefetch path only emits SE/SYS scope, which
// does NOT fill the WGP cache). PF uses the 24-bit immediate offset so all
// tile prefetches share one address register. Addresses stay inside d_ws.
#define PF(p, lit) asm volatile("global_prefetch_b8 %0, off offset:" #lit :: "v"(p))
#define PF0(p)     asm volatile("global_prefetch_b8 %0, off" :: "v"(p))

__device__ __forceinline__ v16bf cat8(v8bf lo, v8bf hi) {
  return __builtin_shufflevector(lo, hi, 0,1,2,3,4,5,6,7,8,9,10,11,12,13,14,15);
}

__device__ __forceinline__ v8f vzero() {
  v8f z;
#pragma unroll
  for (int i = 0; i < 8; ++i) z[i] = 0.f;
  return z;
}

// A fragment (16x32 bf16) from a row-major bf16 row pointer already offset by
// (row*pitch + lh*8): lane holds K {b..b+7} U {b+16..b+23}, b=(lane>>4)*8.
__device__ __forceinline__ v16bf a_frag_bf(const __bf16* prow, int k0) {
  v8bf lo = *(const v8bf*)(prow + k0);
  v8bf hi = *(const v8bf*)(prow + k0 + 16);
  return cat8(lo, hi);
}

// B fragment (32x16 bf16) from a per-lane weight base already offset by
// (n*pitch + lh*8): lane owns column n, K {b..b+7} U {b+16..b+23}, b=k0+(lane>>4)*8.
__device__ __forceinline__ v16bf b_frag_at(const __bf16* p, int off) {
  v8bf lo = *(const v8bf*)(p + off);
  v8bf hi = *(const v8bf*)(p + off + 16);
  return cat8(lo, hi);
}

// A fragment straight from f32 x row (converted to bf16 on the fly).
__device__ __forceinline__ v16bf a_frag_x(const float* xrow, int k0, int lh) {
  int kA = k0 + lh * 8;
  v4f f0 = *(const v4f*)(xrow + kA);
  v4f f1 = *(const v4f*)(xrow + kA + 4);
  v4f f2 = *(const v4f*)(xrow + kA + 16);
  v4f f3 = *(const v4f*)(xrow + kA + 20);
  v16bf a;
#pragma unroll
  for (int i = 0; i < 4; ++i) {
    a[i]      = (__bf16)f0[i];
    a[4 + i]  = (__bf16)f1[i];
    a[8 + i]  = (__bf16)f2[i];
    a[12 + i] = (__bf16)f3[i];
  }
  return a;
}

// K-tail (covers K 672..703): clamp address to stay in-bounds; the clamped
// garbage multiplies zero-padded W1b columns, so the product is exact.
__device__ __forceinline__ v16bf a_frag_x_tail(const float* xrow, int k0, int lh) {
  v16bf a;
  int kA = k0 + lh * 8;
#pragma unroll
  for (int e = 0; e < 8; ++e) {
    int K0 = kA + e;      K0 = (K0 < CC) ? K0 : (CC - 1);
    int K1 = kA + 16 + e; K1 = (K1 < CC) ? K1 : (CC - 1);
    a[e]     = (__bf16)xrow[K0];
    a[8 + e] = (__bf16)xrow[K1];
  }
  return a;
}

__global__ void cvt_pad_kernel(const float* __restrict__ src,
                               __bf16* __restrict__ dst,
                               int srows, int scols, int drows, int dcols) {
  int i = blockIdx.x * blockDim.x + threadIdx.x;
  if (i >= drows * dcols) return;
  int r = i / dcols, k = i - r * dcols;
  float v = (r < srows && k < scols) ? src[(size_t)r * scols + k] : 0.f;
  dst[i] = (__bf16)v;
}

template <bool XB>
__global__ __launch_bounds__(512, 1)
void snn_persistent(const float* __restrict__ x,
                    const __bf16* __restrict__ xb,
                    const __bf16* __restrict__ W1b,
                    const __bf16* __restrict__ Wrecb,
                    const __bf16* __restrict__ W2b,
                    const __bf16* __restrict__ W3b,
                    const float* __restrict__ alpha1,
                    const float* __restrict__ rho1,
                    const float* __restrict__ beta_a1,
                    const float* __restrict__ alpha2,
                    const float* __restrict__ rho2,
                    const float* __restrict__ beta_a2,
                    const float* __restrict__ beta_out,
                    float* __restrict__ out) {
  __shared__ __align__(16) __bf16 s1[16 * H1];  // layer-1 spikes (A matrix), 32 KB
  __shared__ __align__(16) __bf16 s2[16 * H2];  // layer-2 spikes (A matrix), 16 KB

  const int tid  = threadIdx.x;
  const int w    = tid >> 5;
  const int lane = tid & 31;
  const int l15  = lane & 15;
  const int lh   = lane >> 4;
  const int b0   = blockIdx.x * 16;

  for (int i = tid; i < 16 * H1; i += blockDim.x) s1[i] = (__bf16)0.f;
  for (int i = tid; i < 16 * H2; i += blockDim.x) s2[i] = (__bf16)0.f;

  // Per-lane neuron parameters, indexed by owned output column (C-layout N).
  float al1[NT1], rh1[NT1], ba1[NT1];
  const int n1b = w * (NT1 * 16);
#pragma unroll
  for (int n = 0; n < NT1; ++n) {
    int c = n1b + n * 16 + l15;
    al1[n] = alpha1[c]; rh1[n] = rho1[c]; ba1[n] = beta_a1[c];
  }
  float al2[NT2], rh2[NT2], ba2[NT2];
  const int n2b = w * (NT2 * 16);
#pragma unroll
  for (int n = 0; n < NT2; ++n) {
    int c = n2b + n * 16 + l15;
    al2[n] = alpha2[c]; rh2[n] = rho2[c]; ba2[n] = beta_a2[c];
  }
  float bo = 0.f;
  if (w < 3) { int c = w * 16 + l15; bo = beta_out[c < OO ? c : OO - 1]; }

  // ONE per-lane base pointer per weight matrix; the N-tile stride is a
  // compile-time constant that folds into the load's 24-bit immediate offset.
  const __bf16* w1base = W1b   + (size_t)(n1b + l15) * CP + lh * 8;
  const __bf16* wrbase = Wrecb + (size_t)(n1b + l15) * H1 + lh * 8;
  const __bf16* w2base = W2b   + (size_t)(n2b + l15) * H1 + lh * 8;
  const __bf16* w3base = W3b   + (size_t)(w * 16 + l15) * H2 + lh * 8;

  // LDS A-row pointers for this lane (row = l15, K-half lh).
  const __bf16* s1row = s1 + l15 * H1 + lh * 8;
  const __bf16* s2row = s2 + l15 * H2 + lh * 8;

  // Membrane / adaptation state lives in VGPRs in WMMA C-layout.
  v8f v1[NT1], a1[NT1], v2[NT2], a2[NT2], vout, osum;
#pragma unroll
  for (int n = 0; n < NT1; ++n) { v1[n] = vzero(); a1[n] = vzero(); }
#pragma unroll
  for (int n = 0; n < NT2; ++n) { v2[n] = vzero(); a2[n] = vzero(); }
  vout = vzero(); osum = vzero();

  const float*  xrow  = x  + (size_t)(b0 + l15) * TT * CC;            // f32 path
  const __bf16* xrowb = XB ? (xb + (size_t)(b0 + l15) * TT * CP + lh * 8)
                           : (const __bf16*)nullptr;                  // bf16 path
  __syncthreads();

  for (int t = 0; t < TT; ++t) {
    // ---------- layer 1: I1 = xt @ W1^T + spk1 @ Wrec^T ----------
    v8f acc[NT1];
#pragma unroll
    for (int n = 0; n < NT1; ++n) acc[n] = vzero();

    if (XB) {
      const __bf16* xr = xrowb + (size_t)t * CP;
      {  // warm WGP$ with next step's x row while this step computes
        const __bf16* xn = xrowb + (size_t)((t + 1 < TT) ? t + 1 : t) * CP;
#pragma unroll
        for (int j = 0; j < CP / 64; ++j) PF0(xn + j * 64);
      }
      for (int k0 = 0; k0 < CP; k0 += 64) {  // one 128B line per tile per iter
        const __bf16* p1 = w1base + k0;
        PF(p1, 128); PF(p1, 22656); PF(p1, 45184); PF(p1, 67712);
        v16bf a0 = a_frag_bf(xr, k0);
        v16bf a1f = a_frag_bf(xr, k0 + 32);
        v16bf b0[NT1], b1[NT1];
#pragma unroll
        for (int n = 0; n < NT1; ++n) b0[n] = b_frag_at(w1base, n * W1_TILE + k0);
#pragma unroll
        for (int n = 0; n < NT1; ++n) b1[n] = b_frag_at(w1base, n * W1_TILE + k0 + 32);
#pragma unroll
        for (int n = 0; n < NT1; ++n) acc[n] = wmma_bf16(a0, b0[n], acc[n]);
#pragma unroll
        for (int n = 0; n < NT1; ++n) acc[n] = wmma_bf16(a1f, b1[n], acc[n]);
      }
    } else {
      const float* xr = xrow + (size_t)t * CC;
      for (int k0 = 0; k0 < CP - 32; k0 += 32) {
        v16bf a = a_frag_x(xr, k0, lh);
        v16bf b[NT1];
#pragma unroll
        for (int n = 0; n < NT1; ++n) b[n] = b_frag_at(w1base, n * W1_TILE + k0);
#pragma unroll
        for (int n = 0; n < NT1; ++n) acc[n] = wmma_bf16(a, b[n], acc[n]);
      }
      {  // masked K tail (672..703)
        v16bf a = a_frag_x_tail(xr, CP - 32, lh);
        v16bf b[NT1];
#pragma unroll
        for (int n = 0; n < NT1; ++n) b[n] = b_frag_at(w1base, n * W1_TILE + (CP - 32));
#pragma unroll
        for (int n = 0; n < NT1; ++n) acc[n] = wmma_bf16(a, b[n], acc[n]);
      }
    }

    for (int k0 = 0; k0 < H1; k0 += 64) {
      const __bf16* pr = wrbase + k0;
      PF(pr, 128); PF(pr, 32896); PF(pr, 65664); PF(pr, 98432);
      v16bf a0 = a_frag_bf(s1row, k0);
      v16bf a1f = a_frag_bf(s1row, k0 + 32);
      v16bf b0[NT1], b1[NT1];
#pragma unroll
      for (int n = 0; n < NT1; ++n) b0[n] = b_frag_at(wrbase, n * WR_TILE + k0);
#pragma unroll
      for (int n = 0; n < NT1; ++n) b1[n] = b_frag_at(wrbase, n * WR_TILE + k0 + 32);
#pragma unroll
      for (int n = 0; n < NT1; ++n) acc[n] = wmma_bf16(a0, b0[n], acc[n]);
#pragma unroll
      for (int n = 0; n < NT1; ++n) acc[n] = wmma_bf16(a1f, b1[n], acc[n]);
    }
    __syncthreads();  // everyone finished reading spk1(t-1)

    // LIF update + spike write (each (M,N) element owned by exactly one lane)
#pragma unroll
    for (int n = 0; n < NT1; ++n) {
      const int col = n1b + n * 16 + l15;
#pragma unroll
      for (int r = 0; r < 8; ++r) {
        const int M = r + 8 * lh;
        float sOld = (float)s1[M * H1 + col];
        float vv = al1[n] * (v1[n][r] - sOld * SPIKE_TH)
                 + (1.f - al1[n]) * (acc[n][r] - a1[n][r]);
        float s = (vv - SPIKE_TH >= 0.f) ? 1.f : 0.f;
        a1[n][r] = rh1[n] * a1[n][r] + ba1[n] * s;
        v1[n][r] = vv;
        s1[M * H1 + col] = (__bf16)s;
      }
    }
    __syncthreads();  // s1(t) visible to all waves

    // ---------- layer 2: I2 = s1 @ W2^T ----------
    v8f acc2[NT2];
#pragma unroll
    for (int n = 0; n < NT2; ++n) acc2[n] = vzero();
    for (int k0 = 0; k0 < H1; k0 += 64) {
      const __bf16* p2 = w2base + k0;
      PF(p2, 128); PF(p2, 32896);
      v16bf a0 = a_frag_bf(s1row, k0);
      v16bf a1f = a_frag_bf(s1row, k0 + 32);
      v16bf b0[NT2], b1[NT2];
#pragma unroll
      for (int n = 0; n < NT2; ++n) b0[n] = b_frag_at(w2base, n * W2_TILE + k0);
#pragma unroll
      for (int n = 0; n < NT2; ++n) b1[n] = b_frag_at(w2base, n * W2_TILE + k0 + 32);
#pragma unroll
      for (int n = 0; n < NT2; ++n) acc2[n] = wmma_bf16(a0, b0[n], acc2[n]);
#pragma unroll
      for (int n = 0; n < NT2; ++n) acc2[n] = wmma_bf16(a1f, b1[n], acc2[n]);
    }
#pragma unroll
    for (int n = 0; n < NT2; ++n) {
      const int col = n2b + n * 16 + l15;
#pragma unroll
      for (int r = 0; r < 8; ++r) {
        const int M = r + 8 * lh;
        float sOld = (float)s2[M * H2 + col];
        float vv = al2[n] * (v2[n][r] - sOld * SPIKE_TH)
                 + (1.f - al2[n]) * (acc2[n][r] - a2[n][r]);
        float s = (vv - SPIKE_TH >= 0.f) ? 1.f : 0.f;
        a2[n][r] = rh2[n] * a2[n][r] + ba2[n] * s;
        v2[n][r] = vv;
        s2[M * H2 + col] = (__bf16)s;
      }
    }
    __syncthreads();  // s2(t) visible

    // ---------- layer 3 readout (waves 0..2 cover the 48 padded cols) ----------
    if (w < 3) {  // wave-uniform branch: EXEC stays all-ones for WMMA
      v8f acc3 = vzero();
      for (int k0 = 0; k0 < H2; k0 += 64) {
        v16bf a0 = a_frag_bf(s2row, k0);
        v16bf a1f = a_frag_bf(s2row, k0 + 32);
        acc3 = wmma_bf16(a0, b_frag_at(w3base, k0), acc3);
        acc3 = wmma_bf16(a1f, b_frag_at(w3base, k0 + 32), acc3);
      }
#pragma unroll
      for (int r = 0; r < 8; ++r) {
        vout[r] = bo * vout[r] + (1.f - bo) * acc3[r];
        osum[r] += vout[r];
      }
    }
    __syncthreads();  // protect s2/s1 before next step's writes
  }

  if (w < 3) {
    const int col = w * 16 + l15;
    if (col < OO) {
#pragma unroll
      for (int r = 0; r < 8; ++r) {
        const int M = r + 8 * lh;
        out[(size_t)(b0 + M) * OO + col] = osum[r] * (1.f / (float)TT);
      }
    }
  }
}

extern "C" void kernel_launch(void* const* d_in, const int* in_sizes, int n_in,
                              void* d_out, int out_size, void* d_ws, size_t ws_size,
                              hipStream_t stream) {
  (void)in_sizes; (void)n_in; (void)out_size;
  const float* x        = (const float*)d_in[0];
  const float* W1       = (const float*)d_in[1];
  const float* Wrec     = (const float*)d_in[2];
  const float* W2       = (const float*)d_in[3];
  const float* W3       = (const float*)d_in[4];
  const float* alpha1   = (const float*)d_in[5];
  const float* rho1     = (const float*)d_in[6];
  const float* beta_a1  = (const float*)d_in[7];
  const float* alpha2   = (const float*)d_in[8];
  const float* rho2     = (const float*)d_in[9];
  const float* beta_a2  = (const float*)d_in[10];
  const float* beta_out = (const float*)d_in[11];
  float* out = (float*)d_out;

  char* ws = (char*)d_ws;
  __bf16* W1b   = (__bf16*)(ws + WS_W1);    // 1024 x 704
  __bf16* Wrecb = (__bf16*)(ws + WS_WREC);  // 1024 x 1024
  __bf16* W2b   = (__bf16*)(ws + WS_W2);    //  512 x 1024
  __bf16* W3b   = (__bf16*)(ws + WS_W3);    //   48 x  512
  __bf16* Xb    = (__bf16*)(ws + WS_XB);    // (256*250) x 704

  cvt_pad_kernel<<<(1024 * 704 + 255) / 256, 256, 0, stream>>>(W1, W1b, 1024, 700, 1024, 704);
  cvt_pad_kernel<<<(1024 * 1024 + 255) / 256, 256, 0, stream>>>(Wrec, Wrecb, 1024, 1024, 1024, 1024);
  cvt_pad_kernel<<<(512 * 1024 + 255) / 256, 256, 0, stream>>>(W2, W2b, 512, 1024, 512, 1024);
  cvt_pad_kernel<<<(48 * 512 + 255) / 256, 256, 0, stream>>>(W3, W3b, 35, 512, 48, 512);

  if (ws_size >= WS_NEED) {
    // Pre-convert x to zero-padded bf16 [B*T][704]: removes all f32->bf16
    // conversion VALU work (and the masked K-tail) from the persistent loop.
    cvt_pad_kernel<<<(BB * TT * CP + 255) / 256, 256, 0, stream>>>(
        x, Xb, BB * TT, CC, BB * TT, CP);
    snn_persistent<true><<<16, 512, 0, stream>>>(x, Xb, W1b, Wrecb, W2b, W3b,
                                                 alpha1, rho1, beta_a1,
                                                 alpha2, rho2, beta_a2,
                                                 beta_out, out);
  } else {
    snn_persistent<false><<<16, 512, 0, stream>>>(x, nullptr, W1b, Wrecb, W2b, W3b,
                                                  alpha1, rho1, beta_a1,
                                                  alpha2, rho2, beta_a2,
                                                  beta_out, out);
  }
}
